// GNN_15049565405851
// MI455X (gfx1250) — matrix operand order repaired and checked by reading, compiled, not run on hardware
//
#include <hip/hip_runtime.h>
#include <hip/hip_bf16.h>

#define IN_C  128
#define HID_C 64
#define OUT_C 32

typedef __attribute__((ext_vector_type(16))) __bf16 v16bf;
typedef __attribute__((ext_vector_type(8)))  __bf16 v8bf;
typedef __attribute__((ext_vector_type(8)))  float  v8f;

__device__ __forceinline__ v16bf frag_cat(v8bf lo, v8bf hi) {
    union { v16bf v; v8bf h[2]; } u;
    u.h[0] = lo; u.h[1] = hi;
    return u.v;
}

__device__ __forceinline__ void fatomic_add(float* p, float v) {
    // relaxed, device scope -> global_atomic_add_f32 (no CAS loop)
    __hip_atomic_fetch_add(p, v, __ATOMIC_RELAXED, __HIP_MEMORY_SCOPE_AGENT);
}

__device__ __forceinline__ unsigned pack2bf(float a, float b) {
    union { __bf16 b[2]; unsigned u; } p;
    p.b[0] = (__bf16)a; p.b[1] = (__bf16)b;
    return p.u;
}

// ---------------------------------------------------------------- degree prep
__global__ void gcn_init_deg(float* deg, int n) {
    int i = blockIdx.x * blockDim.x + threadIdx.x;
    if (i < n) deg[i] = 1.0f;               // self-loop
}

__global__ void gcn_deg_accum(float* deg, const int* __restrict__ dst, int E) {
    int e = blockIdx.x * blockDim.x + threadIdx.x;
    if (e < E) fatomic_add(&deg[dst[e]], 1.0f);
}

__global__ void gcn_rsqrt(float* deg, int n) {
    int i = blockIdx.x * blockDim.x + threadIdx.x;
    if (i < n) deg[i] = rsqrtf(deg[i]);     // deg >= 1 always (self-loop)
}

// ---------------------------------------------------------------- GEMM1: h = x @ W1
// 128 rows per block, 8 waves x 16 rows, bf16 WMMA f32-accum.
// sX row-major (A frags contiguous b128); W1 staged TRANSPOSED (B frags contiguous b128).
__global__ __launch_bounds__(256) void gcn_gemm1_wmma(const float* __restrict__ x,
                                                      const float* __restrict__ W1,
                                                      float* __restrict__ h, int n) {
    __shared__ __attribute__((aligned(16))) __bf16 sX [128][IN_C + 8];   // stride 272B
    __shared__ __attribute__((aligned(16))) __bf16 sWt[HID_C][IN_C + 8]; // W1^T

    const int t     = threadIdx.x;
    const int mBase = blockIdx.x * 128;

    // x tile: float4 coalesced reads, packed 4x bf16 (b64) LDS stores
    for (int idx = t * 4; idx < 128 * IN_C; idx += 256 * 4) {
        int r = idx >> 7, c = idx & (IN_C - 1);
        int gr = mBase + r;
        float4 v = make_float4(0.f, 0.f, 0.f, 0.f);
        if (gr < n) v = *(const float4*)&x[(size_t)gr * IN_C + c];
        uint2 pk; pk.x = pack2bf(v.x, v.y); pk.y = pack2bf(v.z, v.w);
        *(uint2*)&sX[r][c] = pk;
    }
    // W1 (K=128 x N=64) -> sWt[n][k]; 4 consecutive k per thread -> b64 stores
    for (int chunk = t; chunk < HID_C * (IN_C / 4); chunk += 256) {
        int nn = chunk >> 5;                 // 0..63
        int k0 = (chunk & 31) * 4;           // 0..124
        float w0 = W1[(size_t)(k0 + 0) * HID_C + nn];
        float w1 = W1[(size_t)(k0 + 1) * HID_C + nn];
        float w2 = W1[(size_t)(k0 + 2) * HID_C + nn];
        float w3 = W1[(size_t)(k0 + 3) * HID_C + nn];
        uint2 pk; pk.x = pack2bf(w0, w1); pk.y = pack2bf(w2, w3);
        *(uint2*)&sWt[nn][k0] = pk;
    }
    __syncthreads();

    const int wave = t >> 5;
    const int lane = t & 31;
    const int m0   = wave * 16;
    const int lrow = lane & 15;
    const int lhi  = lane >> 4;              // half-wave select

    // A fragments: per-lane contiguous b128 pairs; hoisted, reused for all N-tiles
    const __bf16* aP = &sX[m0 + lrow][lhi * 8];
    v16bf afr[IN_C / 32];
    #pragma unroll
    for (int kk = 0; kk < IN_C / 32; ++kk)
        afr[kk] = frag_cat(*(const v8bf*)(aP + kk * 32),
                           *(const v8bf*)(aP + kk * 32 + 16));

    const int row0 = mBase + m0 + lhi * 8;   // first of this half-wave's 8 D rows

    #pragma unroll
    for (int nt = 0; nt < HID_C / 16; ++nt) {
        const __bf16* bP = &sWt[nt * 16 + lrow][lhi * 8];
        v8f acc = {};
        #pragma unroll
        for (int kk = 0; kk < IN_C / 32; ++kk) {
            v16bf b = frag_cat(*(const v8bf*)(bP + kk * 32),
                               *(const v8bf*)(bP + kk * 32 + 16));
            acc = __builtin_amdgcn_wmma_f32_16x16x32_bf16(false, afr[kk], false, b,
                                                          (short)0, acc, false, false);
        }
        float* p = h + (size_t)row0 * HID_C + nt * 16 + lrow;
        if (row0 + 8 <= n) {                 // fast path: whole sub-tile in range
            #pragma unroll
            for (int r = 0; r < 8; ++r) p[(size_t)r * HID_C] = acc[r];
        } else {                             // only last block ever takes this
            #pragma unroll
            for (int r = 0; r < 8; ++r)
                if (row0 + r < n) p[(size_t)r * HID_C] = acc[r];
        }
    }
}

// ---------------------------------------------------------------- layer-1 aggregation
__global__ void gcn_init_agg1(float* __restrict__ agg, const float* __restrict__ h,
                              const float* __restrict__ inv, int n) {
    int idx = blockIdx.x * blockDim.x + threadIdx.x;
    if (idx < n * HID_C) {
        int i = idx >> 6;
        float s = inv[i];
        agg[idx] = h[idx] * s * s;           // self-loop message
    }
}

__global__ void gcn_scatter1(float* __restrict__ agg, const float* __restrict__ h,
                             const float* __restrict__ inv,
                             const int* __restrict__ src, const int* __restrict__ dst,
                             int E) {
    int idx = blockIdx.x * blockDim.x + threadIdx.x;
    int e = idx >> 6, f = idx & (HID_C - 1);
    if (e < E) {
        int s = src[e], d = dst[e];
        float nm = inv[s] * inv[d];
        fatomic_add(&agg[(size_t)d * HID_C + f], h[(size_t)s * HID_C + f] * nm);
    }
}

__global__ void gcn_bias_relu(float* __restrict__ agg, const float* __restrict__ b,
                              int total) {
    int idx = blockIdx.x * blockDim.x + threadIdx.x;
    if (idx < total) {
        float v = agg[idx] + b[idx & (HID_C - 1)];
        agg[idx] = v > 0.0f ? v : 0.0f;
    }
}

// ---------------------------------------------------------------- GEMM2: h2 = h1 @ W2
__global__ __launch_bounds__(256) void gcn_gemm2_wmma(const float* __restrict__ h1,
                                                      const float* __restrict__ W2,
                                                      float* __restrict__ h2, int n) {
    __shared__ __attribute__((aligned(16))) __bf16 sH [128][HID_C + 8];  // stride 144B
    __shared__ __attribute__((aligned(16))) __bf16 sWt[OUT_C][HID_C + 8];

    const int t     = threadIdx.x;
    const int mBase = blockIdx.x * 128;

    for (int idx = t * 4; idx < 128 * HID_C; idx += 256 * 4) {
        int r = idx >> 6, c = idx & (HID_C - 1);
        int gr = mBase + r;
        float4 v = make_float4(0.f, 0.f, 0.f, 0.f);
        if (gr < n) v = *(const float4*)&h1[(size_t)gr * HID_C + c];
        uint2 pk; pk.x = pack2bf(v.x, v.y); pk.y = pack2bf(v.z, v.w);
        *(uint2*)&sH[r][c] = pk;
    }
    for (int chunk = t; chunk < OUT_C * (HID_C / 4); chunk += 256) {
        int nn = chunk >> 4;                 // 0..31
        int k0 = (chunk & 15) * 4;           // 0..60
        float w0 = W2[(size_t)(k0 + 0) * OUT_C + nn];
        float w1 = W2[(size_t)(k0 + 1) * OUT_C + nn];
        float w2 = W2[(size_t)(k0 + 2) * OUT_C + nn];
        float w3 = W2[(size_t)(k0 + 3) * OUT_C + nn];
        uint2 pk; pk.x = pack2bf(w0, w1); pk.y = pack2bf(w2, w3);
        *(uint2*)&sWt[nn][k0] = pk;
    }
    __syncthreads();

    const int wave = t >> 5;
    const int lane = t & 31;
    const int m0   = wave * 16;
    const int lrow = lane & 15;
    const int lhi  = lane >> 4;

    const __bf16* aP = &sH[m0 + lrow][lhi * 8];
    v16bf afr[HID_C / 32];
    #pragma unroll
    for (int kk = 0; kk < HID_C / 32; ++kk)
        afr[kk] = frag_cat(*(const v8bf*)(aP + kk * 32),
                           *(const v8bf*)(aP + kk * 32 + 16));

    const int row0 = mBase + m0 + lhi * 8;

    #pragma unroll
    for (int nt = 0; nt < OUT_C / 16; ++nt) {
        const __bf16* bP = &sWt[nt * 16 + lrow][lhi * 8];
        v8f acc = {};
        #pragma unroll
        for (int kk = 0; kk < HID_C / 32; ++kk) {
            v16bf b = frag_cat(*(const v8bf*)(bP + kk * 32),
                               *(const v8bf*)(bP + kk * 32 + 16));
            acc = __builtin_amdgcn_wmma_f32_16x16x32_bf16(false, afr[kk], false, b,
                                                          (short)0, acc, false, false);
        }
        float* p = h2 + (size_t)row0 * OUT_C + nt * 16 + lrow;
        if (row0 + 8 <= n) {
            #pragma unroll
            for (int r = 0; r < 8; ++r) p[(size_t)r * OUT_C] = acc[r];
        } else {
            #pragma unroll
            for (int r = 0; r < 8; ++r)
                if (row0 + r < n) p[(size_t)r * OUT_C] = acc[r];
        }
    }
}

// ---------------------------------------------------------------- layer-2 aggregation (into d_out)
__global__ void gcn_init_out(float* __restrict__ out, const float* __restrict__ h2,
                             const float* __restrict__ inv, const float* __restrict__ b2,
                             int n) {
    int idx = blockIdx.x * blockDim.x + threadIdx.x;
    if (idx < n * OUT_C) {
        int i = idx >> 5, f = idx & (OUT_C - 1);
        float s = inv[i];
        out[idx] = h2[idx] * s * s + b2[f];  // self-loop + bias
    }
}

__global__ void gcn_scatter2(float* __restrict__ out, const float* __restrict__ h2,
                             const float* __restrict__ inv,
                             const int* __restrict__ src, const int* __restrict__ dst,
                             int E) {
    int idx = blockIdx.x * blockDim.x + threadIdx.x;
    int e = idx >> 5, f = idx & (OUT_C - 1);
    if (e < E) {
        int s = src[e], d = dst[e];
        float nm = inv[s] * inv[d];
        fatomic_add(&out[(size_t)d * OUT_C + f], h2[(size_t)s * OUT_C + f] * nm);
    }
}

// ---------------------------------------------------------------- launch
extern "C" void kernel_launch(void* const* d_in, const int* in_sizes, int n_in,
                              void* d_out, int out_size, void* d_ws, size_t ws_size,
                              hipStream_t stream) {
    const float* x  = (const float*)d_in[0];
    const int*   ei = (const int*)  d_in[1];
    const float* W1 = (const float*)d_in[2];
    const float* b1 = (const float*)d_in[3];
    const float* W2 = (const float*)d_in[4];
    const float* b2 = (const float*)d_in[5];
    float* out = (float*)d_out;

    const int n = in_sizes[0] / IN_C;
    const int E = in_sizes[1] / 2;
    const int* src = ei;
    const int* dst = ei + E;

    // workspace layout: inv[n] | h[n*64] | agg1[n*64]   (~52 MB)
    float* inv  = (float*)d_ws;
    float* h    = inv + n;
    float* agg1 = h + (size_t)n * HID_C;
    float* h2   = h;                          // reuse h after layer-1 finishes

    const int T = 256;
    // normalization
    gcn_init_deg <<<(n + T - 1) / T, T, 0, stream>>>(inv, n);
    gcn_deg_accum<<<(E + T - 1) / T, T, 0, stream>>>(inv, dst, E);
    gcn_rsqrt    <<<(n + T - 1) / T, T, 0, stream>>>(inv, n);
    // layer 1
    gcn_gemm1_wmma<<<(n + 127) / 128, T, 0, stream>>>(x, W1, h, n);
    gcn_init_agg1 <<<(n * HID_C + T - 1) / T, T, 0, stream>>>(agg1, h, inv, n);
    gcn_scatter1  <<<(E * HID_C + T - 1) / T, T, 0, stream>>>(agg1, h, inv, src, dst, E);
    gcn_bias_relu <<<(n * HID_C + T - 1) / T, T, 0, stream>>>(agg1, b1, n * HID_C);
    // layer 2
    gcn_gemm2_wmma<<<(n + 127) / 128, T, 0, stream>>>(agg1, W2, h2, n);
    gcn_init_out  <<<(n * OUT_C + T - 1) / T, T, 0, stream>>>(out, h2, inv, b2, n);
    gcn_scatter2  <<<(E * OUT_C + T - 1) / T, T, 0, stream>>>(out, h2, inv, src, dst, E);
}
